// Network_72516227826086
// MI455X (gfx1250) — matrix operand order repaired and checked by reading, not compiled, not run on hardware
//
#include <hip/hip_runtime.h>

typedef __attribute__((ext_vector_type(16))) _Float16 v16h;
typedef __attribute__((ext_vector_type(8)))  _Float16 v8h;
typedef __attribute__((ext_vector_type(8)))  float    v8f;

#define WAVES 8
#define TILE_E (WAVES * 16)

#define SQRT3f      1.7320508075688772f
#define INV_SQRT3f  0.5773502691896258f
#define INV_SQRT6x3 0.7071067811865476f   /* sqrt(3)/sqrt(6) */
#define C_Sf        0.3826834323650898f
#define C_Xf        0.9238795325112867f
#define QNEIGHf     0.25f                 /* 1/sqrt(16) */
#define RS1000f     0.031622776601683794f
#define EMB_STEPf   0.3888888888888889f   /* 3.5/9 */
#define EMB_ISTEPf  2.5714285714285716f
#define EMB_SCALEf  2.8234621965789100f   /* sqrt(10)/1.12 */
#define W1_SCALEf   0.31622776601683794f  /* 1/sqrt(10) */
#define W2_SCALEf   0.1f                  /* 1/sqrt(100) */

/* fast silu/sigmoid: v_exp_f32 + v_rcp_f32 instead of IEEE divide sequence */
__device__ __forceinline__ float siluf(float x) {
  return x * __builtin_amdgcn_rcpf(1.f + __expf(-x));
}
__device__ __forceinline__ float sigf(float x) {
  return __builtin_amdgcn_rcpf(1.f + __expf(-x));
}

/* ---------------- utility kernels ---------------- */

__global__ void k_zero(float* __restrict__ p, long n) {
  long t = (long)blockIdx.x * blockDim.x + threadIdx.x;
  for (; t < n; t += (long)gridDim.x * blockDim.x) p[t] = 0.f;
}

__global__ void k_edge_prep(const float* __restrict__ evec, long E,
                            float* __restrict__ r, float* __restrict__ attr) {
  long e = (long)blockIdx.x * blockDim.x + threadIdx.x;
  if (e >= E) return;
  float v0 = evec[e * 3 + 0], v1 = evec[e * 3 + 1], v2 = evec[e * 3 + 2];
  float rr = sqrtf(v0 * v0 + v1 * v1 + v2 * v2);
  r[e] = rr;
  float u = 2.f * (rr / 3.5f - 1.f);
  float cut = (u > 0.f) ? 0.f : ((u < -1.f) ? 1.f : 0.5f * (1.f - __cosf(3.14159265358979f * u)));
  float inv = cut * SQRT3f * __builtin_amdgcn_rcpf(rr);
  attr[e * 4 + 0] = cut;
  attr[e * 4 + 1] = inv * v0;
  attr[e * 4 + 2] = inv * v1;
  attr[e * 4 + 3] = inv * v2;
}

/* generic irrep linear: out[n, ooff + v*ncomp + i] = scale * sum_u W[u*V+v] * in[n, blockoff(u) + i] */
__global__ void k_lin(const float* __restrict__ in, int istr,
                      const float* __restrict__ W, int U, int V, int ncomp,
                      int off0, int cnt0, int off1,
                      float* __restrict__ out, int ostr, int ooff,
                      float scale, long Nn) {
  long t = (long)blockIdx.x * blockDim.x + threadIdx.x;
  long total = Nn * (long)V * ncomp;
  if (t >= total) return;
  int  i  = (int)(t % ncomp); long t2 = t / ncomp;
  int  v  = (int)(t2 % V);    long n  = t2 / V;
  const float* row = in + n * (long)istr;
  float acc = 0.f;
  for (int u = 0; u < U; u++) {
    int off = (u < cnt0) ? (off0 + u * ncomp) : (off1 + (u - cnt0) * ncomp);
    acc += W[u * V + v] * row[off + i];
  }
  out[n * (long)ostr + ooff + v * ncomp + i] = acc * scale;
}

__global__ void k_combine(float* __restrict__ x2, const float* __restrict__ sc,
                          int dim, int boundary, long Nn) {
  long t = (long)blockIdx.x * blockDim.x + threadIdx.x;
  if (t >= Nn * (long)dim) return;
  int d = (int)(t % dim);
  float f = (d < boundary) ? C_Xf : 1.f;
  x2[t] = C_Sf * sc[t] + f * x2[t];
}

__global__ void k_gate(const float* __restrict__ in, int istr, float* __restrict__ out, int ostr,
                       int n_s, int ng1, int ng2, long Nn) {
  int odim = n_s + 3 * (ng1 + ng2);
  long t = (long)blockIdx.x * blockDim.x + threadIdx.x;
  if (t >= Nn * (long)odim) return;
  int d = (int)(t % odim); long n = t / odim;
  const float* ir = in + n * (long)istr;
  float val;
  if (d < n_s) {
    val = siluf(ir[d]);
  } else {
    int dd = d - n_s;
    int base = n_s + ng1 + ng2;
    if (dd < 3 * ng1) {
      int u = dd / 3;
      val = ir[base + dd] * sigf(ir[n_s + u]);
    } else {
      int dd2 = dd - 3 * ng1;
      int u = dd2 / 3;
      val = ir[base + 3 * ng1 + dd2] * sigf(ir[n_s + ng1 + u]);
    }
  }
  out[n * (long)ostr + d] = val;
}

__global__ void k_final(const float* __restrict__ sc, const float* __restrict__ x2,
                        const int* __restrict__ batch, float* __restrict__ out, long Nn) {
  long n = (long)blockIdx.x * blockDim.x + threadIdx.x;
  if (n >= Nn) return;
  float v = (C_Sf * sc[n] + C_Xf * x2[n]) * RS1000f;
  unsafeAtomicAdd(&out[batch[n]], v);
}

/* ---------------- fused per-edge conv kernel (radial MLP via WMMA + TP + scatter) ------------- */
/* LDS layout: lw1 [112][32] f16 (n-major)   : padded fc_w1^T * (1/sqrt10)
 *             lw2 [NUMEL][128] f16 (n-major): padded fc_w2^T * (1/sqrt100)
 *             htile [WAVES][16][128] f16    : silu hidden activations (row = edge in tile)
 *             wtile [WAVES][16][NUMEL] f32  : per-edge TP weights                               */
template <int CONV, int NUMEL, int XXDIM, int MIDDIM, int MULT>
__launch_bounds__(256)
__global__ void k_conv_edges(const float* __restrict__ r, const float* __restrict__ attr,
                             const float* __restrict__ xx,
                             const int* __restrict__ esrc, const int* __restrict__ edst,
                             const float* __restrict__ fcw1, const float* __restrict__ fcw2,
                             float* __restrict__ agg, long En) {
  extern __shared__ char smem[];
  _Float16* lw1   = (_Float16*)smem;                    /* 112*32 halves    */
  _Float16* lw2   = lw1 + 112 * 32;                     /* NUMEL*128 halves */
  _Float16* htile = lw2 + NUMEL * 128;                  /* WAVES*16*128     */
  float*    wtile = (float*)(htile + WAVES * 16 * 128); /* WAVES*16*NUMEL   */

  const int tid = threadIdx.x;

  /* early prefetch of the TP-phase gather row (hides L2 latency behind the WMMA work) */
  {
    long ep = (long)blockIdx.x * TILE_E + (tid >> 1);
    if (ep < En) {
      int s = esrc[ep];
      __builtin_prefetch(xx + (long)s * XXDIM, 0, 3);
      __builtin_prefetch(xx + (long)s * XXDIM + 32, 0, 3);
    }
  }

  /* stage + convert weights into LDS (n-major so B-fragments are contiguous 16-half runs) */
  for (int idx = tid; idx < 112 * 32; idx += 256) {
    int n = idx >> 5, k = idx & 31;
    float v = (k < 10 && n < 100) ? fcw1[k * 100 + n] * W1_SCALEf : 0.f;
    lw1[idx] = (_Float16)v;
  }
  for (int idx = tid; idx < NUMEL * 128; idx += 256) {
    int n = idx >> 7, k = idx & 127;
    float v = (k < 100) ? fcw2[k * NUMEL + n] * W2_SCALEf : 0.f;
    lw2[idx] = (_Float16)v;
  }
  __syncthreads();

  const int wv = tid >> 5, ln = tid & 31;
  const int m = ln & 15, hi = ln >> 4;     /* A/B: row/col = lane%16; K-half select = lane/16 */
  const long e0 = (long)blockIdx.x * TILE_E + wv * 16;
  const long em = e0 + m;
  const float rv = (em < En) ? r[em] : 1.0e9f;

  /* A1 fragment: 16 edges x K=32 radial embedding, built in-register.
   * k order used here for A and B: half i -> K = (lane/16)*16 + i */
  v16h a1;
#pragma unroll
  for (int i = 0; i < 16; i++) {
    int v = i >> 1, hf = i & 1;
    int K = (hi << 4) + i; (void)v; (void)hf;
    float val = 0.f;
    if (K < 10) {
      float d = (rv - (float)K * EMB_STEPf) * EMB_ISTEPf;
      val = __expf(-d * d) * EMB_SCALEf;
    }
    a1[i] = (_Float16)val;
  }

  _Float16* ht = htile + wv * (16 * 128);
  /* zero hidden pad columns 112..127 */
#pragma unroll
  for (int i = 0; i < 8; i++) {
    int idx = ln * 8 + i;
    ht[(idx >> 4) * 128 + 112 + (idx & 15)] = (_Float16)0.f;
  }

  /* WMMA1: (16 x 32) @ (32 x 16) per hidden tile; silu; store f16 to htile (row-major by edge) */
  for (int nt = 0; nt < 7; nt++) {
    v16h b = *(const v16h*)(lw1 + (nt * 16 + m) * 32 + hi * 16);
    v8f c = {};
    c = __builtin_amdgcn_wmma_f32_16x16x32_f16(false, a1, false, b, (short)0, c, false, false);
#pragma unroll
    for (int g = 0; g < 8; g++) {
      float h = siluf(c[g]);
      ht[(g + hi * 8) * 128 + nt * 16 + m] = (_Float16)h;  /* C/D: M=g+8*hi, N=lane%16 */
    }
  }
  __syncthreads();

  /* WMMA2: (16 x 128) @ (128 x 16) per numel tile; write f32 w-tile to LDS */
  float* wt = wtile + wv * (16 * NUMEL);
  constexpr int NT2 = NUMEL / 16;
  for (int nt2 = 0; nt2 < NT2; nt2++) {
    v8f acc = {};
    for (int kc = 0; kc < 4; kc++) {
      v8h lo  = *(const v8h*)(ht + m * 128 + kc * 32 + hi * 16);
      v8h hiv = *(const v8h*)(ht + m * 128 + kc * 32 + hi * 16 + 8);
      v16h a2;
#pragma unroll
      for (int i = 0; i < 8; i++) { a2[i] = lo[i]; a2[8 + i] = hiv[i]; }
      v16h b2 = *(const v16h*)(lw2 + (nt2 * 16 + m) * 128 + kc * 32 + hi * 16);
      acc = __builtin_amdgcn_wmma_f32_16x16x32_f16(false, a2, false, b2, (short)0, acc, false, false);
    }
#pragma unroll
    for (int g = 0; g < 8; g++) wt[(g + hi * 8) * NUMEL + nt2 * 16 + m] = acc[g];
  }
  __syncthreads();

  /* tensor product + atomic scatter: 2 threads per edge, MULT/2 channels each */
  const int eloc = tid >> 1, halfc = tid & 1;
  const long e = (long)blockIdx.x * TILE_E + eloc;
  if (e < En) {
    const int srcn = esrc[e], dstn = edst[e];
    const float* xr = xx + (long)srcn * XXDIM;
    float* ar = agg + (long)dstn * MIDDIM;
    const float sh0 = attr[e * 4 + 0];
    const float s1 = attr[e * 4 + 1], s2 = attr[e * 4 + 2], s3 = attr[e * 4 + 3];
    const float* wr = wtile + (eloc >> 4) * (16 * NUMEL) + (eloc & 15) * NUMEL;
    const int u0 = halfc * (MULT / 2);
    for (int uu = 0; uu < MULT / 2; uu++) {
      const int u = u0 + uu;
      if constexpr (CONV == 1) {
        float x0 = xr[u];
        float w0 = wr[u], w1 = wr[16 + u];
        unsafeAtomicAdd(ar + u, QNEIGHf * x0 * sh0 * w0);
        float c1 = QNEIGHf * SQRT3f * x0 * w1;
        unsafeAtomicAdd(ar + 16 + u * 3 + 0, c1 * s1);
        unsafeAtomicAdd(ar + 16 + u * 3 + 1, c1 * s2);
        unsafeAtomicAdd(ar + 16 + u * 3 + 2, c1 * s3);
      } else if constexpr (CONV == 2) {
        float x0 = xr[u];
        float vx = xr[32 + u * 3 + 0], vy = xr[32 + u * 3 + 1], vz = xr[32 + u * 3 + 2];
        float w0 = wr[u], w1 = wr[32 + u], w2 = wr[64 + u], w3 = wr[96 + u], w4 = wr[128 + u];
        unsafeAtomicAdd(ar + u, QNEIGHf * x0 * sh0 * w0);
        float c1 = QNEIGHf * SQRT3f * x0 * w1;
        unsafeAtomicAdd(ar + 32 + u * 3 + 0, c1 * s1);
        unsafeAtomicAdd(ar + 32 + u * 3 + 1, c1 * s2);
        unsafeAtomicAdd(ar + 32 + u * 3 + 2, c1 * s3);
        float c2 = QNEIGHf * SQRT3f * sh0 * w2;
        unsafeAtomicAdd(ar + 128 + u * 3 + 0, c2 * vx);
        unsafeAtomicAdd(ar + 128 + u * 3 + 1, c2 * vy);
        unsafeAtomicAdd(ar + 128 + u * 3 + 2, c2 * vz);
        float dot = vx * s1 + vy * s2 + vz * s3;
        unsafeAtomicAdd(ar + 224 + u, QNEIGHf * INV_SQRT3f * dot * w3);
        float cx = vy * s3 - vz * s2, cy = vz * s1 - vx * s3, cz = vx * s2 - vy * s1;
        float c4 = QNEIGHf * INV_SQRT6x3 * w4;
        unsafeAtomicAdd(ar + 256 + u * 3 + 0, c4 * cx);
        unsafeAtomicAdd(ar + 256 + u * 3 + 1, c4 * cy);
        unsafeAtomicAdd(ar + 256 + u * 3 + 2, c4 * cz);
      } else { /* CONV == 3 */
        float x0 = xr[u];
        float vx = xr[32 + u * 3 + 0], vy = xr[32 + u * 3 + 1], vz = xr[32 + u * 3 + 2];
        float w0 = wr[u], w1 = wr[32 + u];
        unsafeAtomicAdd(ar + u, QNEIGHf * x0 * sh0 * w0);
        float dot = vx * s1 + vy * s2 + vz * s3;
        unsafeAtomicAdd(ar + 32 + u, QNEIGHf * INV_SQRT3f * dot * w1);
      }
    }
  }
}

/* ---------------- host side ---------------- */

static inline int nblk(long total) { return (int)((total + 255) / 256); }

extern "C" void kernel_launch(void* const* d_in, const int* in_sizes, int n_in,
                              void* d_out, int out_size, void* d_ws, size_t ws_size,
                              hipStream_t stream) {
  const long N = in_sizes[0] / 16;   /* x: (N,16)        */
  const long E = in_sizes[1] / 3;    /* edge_vec: (E,3)  */

  const float* x    = (const float*)d_in[0];
  const float* evec = (const float*)d_in[1];

  /* param leaves in jax-pytree (sorted-key) order */
  static const long leaf_sz[22] = {1000, 3200, 256, 1024, 512, 1024,
                                   1000, 16000, 1024, 1024, 6144, 2048, 1024, 3072, 1024,
                                   1000, 6400, 1024, 1024, 1024, 64, 32};
  const float* P[22];
  const int *esrc, *edst, *batch;
  if (n_in >= 27) {
    for (int i = 0; i < 22; i++) P[i] = (const float*)d_in[2 + i];
    esrc = (const int*)d_in[24]; edst = (const int*)d_in[25]; batch = (const int*)d_in[26];
  } else {
    const float* base = (const float*)d_in[2];
    long off = 0;
    for (int i = 0; i < 22; i++) { P[i] = base + off; off += leaf_sz[i]; }
    esrc = (const int*)d_in[3]; edst = (const int*)d_in[4]; batch = (const int*)d_in[5];
  }

  /* workspace partition (floats) */
  float* w = (float*)d_ws;
  float* r     = w;  w += E;
  float* attr  = w;  w += 4 * E;
  float* nodeX = w;  w += N * 224;
  float* xxb   = w;  w += N * 288;   /* doubles as xx and x2/pre-gate buffer per conv */
  float* scb   = w;  w += N * 288;
  float* agg   = w;  w += N * 352;
  (void)ws_size;

  const float rs16 = 0.25f, rs32 = 0.17677669529663689f, rs64 = 0.125f;
  const int egrid = (int)((E + TILE_E - 1) / TILE_E);
  const int smem1 = 112 * 32 * 2 + 32 * 128 * 2 + WAVES * 16 * 128 * 2 + WAVES * 16 * 32 * 4;
  const int smem2 = 112 * 32 * 2 + 160 * 128 * 2 + WAVES * 16 * 128 * 2 + WAVES * 16 * 160 * 4;
  const int smem3 = 112 * 32 * 2 + 64 * 128 * 2 + WAVES * 16 * 128 * 2 + WAVES * 16 * 64 * 4;
  (void)hipFuncSetAttribute(reinterpret_cast<const void*>(k_conv_edges<1, 32, 16, 64, 16>),
                            hipFuncAttributeMaxDynamicSharedMemorySize, smem1);
  (void)hipFuncSetAttribute(reinterpret_cast<const void*>(k_conv_edges<2, 160, 128, 352, 32>),
                            hipFuncAttributeMaxDynamicSharedMemorySize, smem2);
  (void)hipFuncSetAttribute(reinterpret_cast<const void*>(k_conv_edges<3, 64, 224, 64, 32>),
                            hipFuncAttributeMaxDynamicSharedMemorySize, smem3);

  k_edge_prep<<<nblk(E), 256, 0, stream>>>(evec, E, r, attr);

  /* ---------- conv1: in = x (dim 16) ---------- */
  k_lin<<<nblk(N * 16), 256, 0, stream>>>(x, 16, P[2], 16, 16, 1, 0, 16, 0, xxb, 16, 0, rs16, N);
  k_zero<<<nblk(N * 160), 256, 0, stream>>>(scb, N * 160);
  k_lin<<<nblk(N * 64), 256, 0, stream>>>(x, 16, P[5], 16, 64, 1, 0, 16, 0, scb, 160, 0, rs16, N);
  k_zero<<<nblk(N * 64), 256, 0, stream>>>(agg, N * 64);
  k_conv_edges<1, 32, 16, 64, 16><<<egrid, 256, smem1, stream>>>(r, attr, xxb, esrc, edst, P[0], P[1], agg, E);
  k_lin<<<nblk(N * 64), 256, 0, stream>>>(agg, 64, P[3], 16, 64, 1, 0, 16, 0, xxb, 160, 0, rs16, N);
  k_lin<<<nblk(N * 96), 256, 0, stream>>>(agg, 64, P[4], 16, 32, 3, 16, 16, 0, xxb, 160, 64, rs16, N);
  k_combine<<<nblk(N * 160), 256, 0, stream>>>(xxb, scb, 160, 64, N);
  k_gate<<<nblk(N * 128), 256, 0, stream>>>(xxb, 160, nodeX, 128, 32, 32, 0, N);

  /* ---------- conv2: in = nodeX (dim 128) ---------- */
  k_lin<<<nblk(N * 32), 256, 0, stream>>>(nodeX, 128, P[8], 32, 32, 1, 0, 32, 0, xxb, 128, 0, rs32, N);
  k_lin<<<nblk(N * 96), 256, 0, stream>>>(nodeX, 128, P[9], 32, 32, 3, 32, 32, 0, xxb, 128, 32, rs32, N);
  k_zero<<<nblk(N * 288), 256, 0, stream>>>(scb, N * 288);
  k_lin<<<nblk(N * 96), 256, 0, stream>>>(nodeX, 128, P[13], 32, 96, 1, 0, 32, 0, scb, 288, 0, rs32, N);
  k_lin<<<nblk(N * 96), 256, 0, stream>>>(nodeX, 128, P[14], 32, 32, 3, 32, 32, 0, scb, 288, 96, rs32, N);
  k_zero<<<nblk(N * 352), 256, 0, stream>>>(agg, N * 352);
  k_conv_edges<2, 160, 128, 352, 32><<<egrid, 256, smem2, stream>>>(r, attr, xxb, esrc, edst, P[6], P[7], agg, E);
  k_lin<<<nblk(N * 96), 256, 0, stream>>>(agg, 352, P[10], 64, 96, 1, 0, 32, 224, xxb, 288, 0, rs64, N);
  k_lin<<<nblk(N * 96), 256, 0, stream>>>(agg, 352, P[11], 64, 32, 3, 32, 32, 128, xxb, 288, 96, rs64, N);
  k_lin<<<nblk(N * 96), 256, 0, stream>>>(agg, 352, P[12], 32, 32, 3, 256, 32, 0, xxb, 288, 192, rs32, N);
  k_combine<<<nblk(N * 288), 256, 0, stream>>>(xxb, scb, 288, 192, N);
  k_gate<<<nblk(N * 224), 256, 0, stream>>>(xxb, 288, nodeX, 224, 32, 32, 32, N);

  /* ---------- conv3: in = nodeX (dim 224) ---------- */
  k_lin<<<nblk(N * 32), 256, 0, stream>>>(nodeX, 224, P[17], 32, 32, 1, 0, 32, 0, xxb, 224, 0, rs32, N);
  k_lin<<<nblk(N * 96), 256, 0, stream>>>(nodeX, 224, P[18], 32, 32, 3, 32, 32, 0, xxb, 224, 32, rs32, N);
  k_lin<<<nblk(N * 96), 256, 0, stream>>>(nodeX, 224, P[19], 32, 32, 3, 128, 32, 0, xxb, 224, 128, rs32, N);
  k_lin<<<nblk(N * 1), 256, 0, stream>>>(nodeX, 224, P[21], 32, 1, 1, 0, 32, 0, scb, 1, 0, rs32, N);
  k_zero<<<nblk(N * 64), 256, 0, stream>>>(agg, N * 64);
  k_conv_edges<3, 64, 224, 64, 32><<<egrid, 256, smem3, stream>>>(r, attr, xxb, esrc, edst, P[15], P[16], agg, E);
  k_lin<<<nblk(N * 1), 256, 0, stream>>>(agg, 64, P[20], 64, 1, 1, 0, 32, 32, xxb, 1, 0, rs64, N);

  /* ---------- readout ---------- */
  k_zero<<<1, 256, 0, stream>>>((float*)d_out, out_size);
  k_final<<<nblk(N), 256, 0, stream>>>(scb, xxb, batch, (float*)d_out, N);
}
